// InfiniAttention_75591424410248
// MI455X (gfx1250) — matrix operand order repaired and checked by reading, not compiled, not run on hardware
//
#include <hip/hip_runtime.h>

#define E_DIM 1024
#define S_DIM 2048
#define B_DIM 8
#define H_DIM 16
#define NSEG  16
#define SEG   64
#define DH    64

typedef __bf16 bf16_t;
typedef __attribute__((ext_vector_type(16))) __bf16 v16bf;
typedef __attribute__((ext_vector_type(8)))  __bf16 v8bf;
typedef __attribute__((ext_vector_type(4)))  __bf16 v4bf;
typedef __attribute__((ext_vector_type(8)))  float  v8f;
typedef __attribute__((ext_vector_type(4)))  float  v4f;

// Native gfx1250 bf16 converts (v_cvt_pk_bf16_f32 / shift) instead of bit tricks
__device__ __forceinline__ bf16_t f2bf(float f)  { return (bf16_t)f; }
__device__ __forceinline__ float  bf2f(bf16_t b) { return (float)b; }

__device__ __forceinline__ v16bf cat16(v8bf lo, v8bf hi) {
  v16bf r;
#pragma unroll
  for (int i = 0; i < 8; ++i) { r[i] = lo[i]; r[i + 8] = hi[i]; }
  return r;
}
// A operand (16xK tile, row-major, LDS stride 72 halves). Lane<16: K {k0..k0+7, k0+16..k0+23}
__device__ __forceinline__ v16bf ld_a72(const bf16_t* buf, int row0, int k0, int lane) {
  const int hi = (lane >> 4) & 1, l = lane & 15;
  const bf16_t* p = buf + (row0 + l) * 72 + k0 + hi * 8;
  return cat16(*(const v8bf*)p, *(const v8bf*)(p + 16));
}
// B operand: lane n = row of W (= column of B). Lane<16: K {k0..k0+7, k0+16..k0+23}
__device__ __forceinline__ v16bf ld_b72(const bf16_t* buf, int row0, int k0, int lane) {
  const int hi = (lane >> 4) & 1, l = lane & 15;
  const bf16_t* p = buf + (row0 + l) * 72 + k0 + hi * 8;
  return cat16(*(const v8bf*)p, *(const v8bf*)(p + 16));
}
__device__ __forceinline__ v16bf elu_p1(v16bf v) {
  v16bf r;
#pragma unroll
  for (int i = 0; i < 16; ++i) {
    float f = bf2f(v[i]);
    f = (f > 0.f) ? (f + 1.f) : __expf(f);
    r[i] = f2bf(f);
  }
  return r;
}

// ---------------------------------------------------------------------------
// C[b] = A[b] (MxK, row-major, lda) @ W^T (W is NxK row-major, lda) + bias[N]
// 128x128 block tile, BK=32, 256 threads = 8 waves, each wave 2x4 WMMA subtiles.
// Register-pipelined: next K-tile global loads overlap WMMA compute.
// ---------------------------------------------------------------------------
#define GBM 128
#define GBN 128
#define GBK 32
#define GSTR 40   // halves; pads 32-wide rows, keeps 16B alignment of operand chunks

__global__ __launch_bounds__(256)
void wmma_gemm_abt_bias(const float* __restrict__ A, long long strideA,
                        const float* __restrict__ W,
                        const float* __restrict__ bias,
                        float* __restrict__ C, long long strideC,
                        int lda, int ldc, int K)
{
  __shared__ __align__(16) bf16_t Asm[GBM * GSTR];
  __shared__ __align__(16) bf16_t Bsm[GBN * GSTR];

  const int tid  = threadIdx.x;
  const int lane = tid & 31;
  const int wave = tid >> 5;
  const int hi   = (lane >> 4) & 1;
  const int l15  = lane & 15;
  const int waveM = wave >> 1;   // 0..3 : 32-row strip
  const int waveN = wave & 1;    // 0..1 : 64-col strip

  const int m0 = blockIdx.y * GBM;
  const int n0 = blockIdx.x * GBN;
  const float* Ab = A + (long long)blockIdx.z * strideA;
  float*       Cb = C + (long long)blockIdx.z * strideC;

  // per-thread slice of the cooperative tile load: 4 float4 chunks per matrix
  int rowL[4], kcL[4];
#pragma unroll
  for (int c = 0; c < 4; ++c) {
    const int idx = tid + 256 * c;    // 1024 float4 chunks per tile
    rowL[c] = idx >> 3;               // 8 chunks per 32-wide row
    kcL[c]  = (idx & 7) << 2;
  }

  v8f acc[2][4] = {};
  v4f avR[4], wvR[4];

  // prologue: load K-tile 0 into registers
#pragma unroll
  for (int c = 0; c < 4; ++c) {
    avR[c] = *(const v4f*)(Ab + (long long)(m0 + rowL[c]) * lda + kcL[c]);
    wvR[c] = *(const v4f*)(W  + (long long)(n0 + rowL[c]) * lda + kcL[c]);
  }

  for (int kt = 0; kt < K; kt += GBK) {
    // store staged registers -> bf16 LDS tiles
#pragma unroll
    for (int c = 0; c < 4; ++c) {
      v4bf ap, wp;
#pragma unroll
      for (int i = 0; i < 4; ++i) { ap[i] = f2bf(avR[c][i]); wp[i] = f2bf(wvR[c][i]); }
      *(v4bf*)(&Asm[rowL[c] * GSTR + kcL[c]]) = ap;
      *(v4bf*)(&Bsm[rowL[c] * GSTR + kcL[c]]) = wp;
    }
    __syncthreads();

    // issue next K-tile global loads (overlap with WMMA below)
    const int ktn = kt + GBK;
    if (ktn < K) {
#pragma unroll
      for (int c = 0; c < 4; ++c) {
        const float* pa = Ab + (long long)(m0 + rowL[c]) * lda + ktn + kcL[c];
        const float* pw = W  + (long long)(n0 + rowL[c]) * lda + ktn + kcL[c];
        __builtin_prefetch(pa + GBK, 0, 1);
        __builtin_prefetch(pw + GBK, 0, 1);
        avR[c] = *(const v4f*)pa;
        wvR[c] = *(const v4f*)pw;
      }
    }

    // WMMA compute on the LDS tile
    v16bf a0, a1, bb[4];
    {
      const bf16_t* p0 = &Asm[(waveM * 32 + 0 * 16 + l15) * GSTR + hi * 8];
      a0 = cat16(*(const v8bf*)p0, *(const v8bf*)(p0 + 16));
      const bf16_t* p1 = &Asm[(waveM * 32 + 1 * 16 + l15) * GSTR + hi * 8];
      a1 = cat16(*(const v8bf*)p1, *(const v8bf*)(p1 + 16));
#pragma unroll
      for (int j = 0; j < 4; ++j) {
        const bf16_t* pb = &Bsm[(waveN * 64 + j * 16 + l15) * GSTR + hi * 8];
        bb[j] = cat16(*(const v8bf*)pb, *(const v8bf*)(pb + 16));
      }
    }
#pragma unroll
    for (int j = 0; j < 4; ++j) {
      acc[0][j] = __builtin_amdgcn_wmma_f32_16x16x32_bf16(false, a0, false, bb[j],
                                                          (short)0, acc[0][j], false, false);
      acc[1][j] = __builtin_amdgcn_wmma_f32_16x16x32_bf16(false, a1, false, bb[j],
                                                          (short)0, acc[1][j], false, false);
    }
    __syncthreads();
  }

  // write C (+bias). C layout: VGPR r -> M = r + hi*8, N = lane&15
#pragma unroll
  for (int i = 0; i < 2; ++i) {
#pragma unroll
    for (int j = 0; j < 4; ++j) {
      const int gm = m0 + waveM * 32 + i * 16 + hi * 8;
      const int gn = n0 + waveN * 64 + j * 16 + l15;
      const float bv = bias[gn];
#pragma unroll
      for (int r = 0; r < 8; ++r)
        Cb[(long long)(gm + r) * ldc + gn] = acc[i][j][r] + bv;
    }
  }
}

// ---------------------------------------------------------------------------
// Segmented Infini-attention: one workgroup per (b,h). The torch-style reshape
// makes each (h,seg) q/k/v tile a contiguous 64x64 block of q[b] (flat E*E).
// Per segment: scores=qs@ks^T, A_mem=(elu(qs)+1)@mem, softmax, A_dot=P@vs,
// attn = g*A_mem/denom + (1-g)*A_dot, mem += sk^T@vs, z += colsum(sk).
// mem accumulated in persistent per-wave f32 WMMA accumulators; bf16 copy in
// LDS serves as B operand for the next segment's A_mem.
// ---------------------------------------------------------------------------
__global__ __launch_bounds__(256)
void infini_attention_seg(const float* __restrict__ q,
                          const float* __restrict__ k,
                          const float* __restrict__ v,
                          const float* __restrict__ beta,
                          float* __restrict__ attn)
{
  // qs+ks region (2*9216B) is aliased by the fp32 scores buffer (64*68*4=17408B)
  __shared__ __align__(16) char qk_mem[2 * 64 * 72 * sizeof(bf16_t)];
  __shared__ __align__(16) bf16_t skT_s[64 * 72];
  __shared__ __align__(16) bf16_t vsT_s[64 * 72];
  __shared__ __align__(16) bf16_t memT_s[64 * 72];
  __shared__ __align__(16) bf16_t P_s[64 * 72];
  __shared__ float rowsum_s[64];
  __shared__ float z_s[64];

  bf16_t* qs_s = (bf16_t*)qk_mem;
  bf16_t* ks_s = (bf16_t*)(qk_mem + 64 * 72 * sizeof(bf16_t));
  float*  scores_s = (float*)qk_mem;   // stride 68 floats

  const int tid  = threadIdx.x;
  const int lane = tid & 31;
  const int wave = tid >> 5;
  const int hi   = (lane >> 4) & 1;
  const int l15  = lane & 15;
  const int mi   = wave >> 1;          // 0..3 : 16-row (M) subtile
  const int njB  = (wave & 1) * 2;     // N subtile pair base

  const int b = blockIdx.x >> 4;
  const int h = blockIdx.x & 15;

  const float g   = 1.f / (1.f + __expf(-beta[h] * 10.f));
  const float omg = 1.f - g;

  for (int i = tid; i < 64 * 72; i += 256) memT_s[i] = f2bf(0.f);
  if (tid < 64) { z_s[tid] = 0.f; rowsum_s[tid] = 0.f; }
  __syncthreads();

  v8f memAcc[2] = {};

  const int srow = tid >> 2;          // 0..63 row handled by this thread in load phase
  const int dp   = (tid & 3) << 4;    // 16-wide column slice

  for (int seg = 0; seg < NSEG; ++seg) {
    const long long base = (long long)b * (E_DIM * E_DIM)
                         + (long long)(h * NSEG + seg) * (SEG * DH);
    const float* qseg = q + base;
    const float* kseg = k + base;
    const float* vseg = v + base;
    float*       aseg = attn + base;

    // ---- load + elementwise transforms ----
    float rs = 0.f;
#pragma unroll
    for (int c = 0; c < 4; ++c) {
      v4f qv = *(const v4f*)(qseg + srow * DH + dp + c * 4);
      v4f kv = *(const v4f*)(kseg + srow * DH + dp + c * 4);
      v4f vv = *(const v4f*)(vseg + srow * DH + dp + c * 4);
      v4bf qp;
#pragma unroll
      for (int i = 0; i < 4; ++i) {
        const int d = dp + c * 4 + i;
        const float qf = qv[i], kf = kv[i];
        qp[i] = f2bf(qf);
        rs += (qf > 0.f) ? (qf + 1.f) : __expf(qf);                 // rowsum(elu(q)+1) fp32
        ks_s[srow * 72 + d]  = f2bf(kf);
        skT_s[d * 72 + srow] = f2bf((kf > 0.f) ? (kf + 1.f) : __expf(kf));
        vsT_s[d * 72 + srow] = f2bf(vv[i]);
      }
      *(v4bf*)(&qs_s[srow * 72 + dp + c * 4]) = qp;
    }
    atomicAdd(&rowsum_s[srow], rs);
    __syncthreads();

    // ---- pass 1+2: scores = qs@ks^T and A_mem = sq@mem (mem from prev seg) ----
    v8f Sacc[2] = {};
    v8f Macc[2] = {};
#pragma unroll
    for (int k0 = 0; k0 < 64; k0 += 32) {
      v16bf aq  = ld_a72(qs_s, mi * 16, k0, lane);
      v16bf asq = elu_p1(aq);
#pragma unroll
      for (int j = 0; j < 2; ++j) {
        v16bf bk_ = ld_b72(ks_s, (njB + j) * 16, k0, lane);
        Sacc[j] = __builtin_amdgcn_wmma_f32_16x16x32_bf16(false, aq, false, bk_,
                                                          (short)0, Sacc[j], false, false);
        v16bf bm  = ld_b72(memT_s, (njB + j) * 16, k0, lane);
        Macc[j] = __builtin_amdgcn_wmma_f32_16x16x32_bf16(false, asq, false, bm,
                                                          (short)0, Macc[j], false, false);
      }
    }
    __syncthreads();   // all reads of qs/ks/memT complete -> safe to alias scores

    // ---- scores -> fp32 LDS (aliased over qs/ks) ----
#pragma unroll
    for (int j = 0; j < 2; ++j)
#pragma unroll
      for (int r = 0; r < 8; ++r)
        scores_s[(mi * 16 + r + hi * 8) * 68 + (njB + j) * 16 + l15] = Sacc[j][r];
    __syncthreads();

    // ---- causal softmax rows -> P (bf16 A operand) ----
    if (tid < 64) {
      const int sr = tid;
      float mx = -3.402823466e38f;
      for (int t = 0; t <= sr; ++t)
        mx = fmaxf(mx, scores_s[sr * 68 + t] * 0.125f);
      float sum = 0.f;
      for (int t = 0; t <= sr; ++t) {
        float e = __expf(scores_s[sr * 68 + t] * 0.125f - mx);
        scores_s[sr * 68 + t] = e;
        sum += e;
      }
      const float inv = 1.f / sum;
      for (int t = 0; t < 64; ++t)
        P_s[sr * 72 + t] = (t <= sr) ? f2bf(scores_s[sr * 68 + t] * inv) : f2bf(0.f);
    }
    __syncthreads();

    // ---- pass 3: A_dot = P@vs, combine, write attn ----
    v8f Dacc[2] = {};
#pragma unroll
    for (int k0 = 0; k0 < 64; k0 += 32) {
      v16bf ap = ld_a72(P_s, mi * 16, k0, lane);
#pragma unroll
      for (int j = 0; j < 2; ++j) {
        v16bf bvt = ld_b72(vsT_s, (njB + j) * 16, k0, lane);
        Dacc[j] = __builtin_amdgcn_wmma_f32_16x16x32_bf16(false, ap, false, bvt,
                                                          (short)0, Dacc[j], false, false);
      }
    }
#pragma unroll
    for (int j = 0; j < 2; ++j) {
      const int e  = (njB + j) * 16 + l15;
      const float ze = z_s[e];
#pragma unroll
      for (int r = 0; r < 8; ++r) {
        const int sr = mi * 16 + r + hi * 8;
        const float denom = rowsum_s[sr] * ze + 1e-6f;
        aseg[sr * DH + e] = g * (Macc[j][r] / denom) + omg * Dacc[j][r];
      }
    }

    // ---- pass 4: mem^T += vs^T @ sk (persistent f32 accumulators) ----
#pragma unroll
    for (int k0 = 0; k0 < 64; k0 += 32) {
      v16bf av_ = ld_a72(vsT_s, mi * 16, k0, lane);
#pragma unroll
      for (int j = 0; j < 2; ++j) {
        v16bf bsk = ld_b72(skT_s, (njB + j) * 16, k0, lane);
        memAcc[j] = __builtin_amdgcn_wmma_f32_16x16x32_bf16(false, av_, false, bsk,
                                                            (short)0, memAcc[j], false, false);
      }
    }
#pragma unroll
    for (int j = 0; j < 2; ++j)
#pragma unroll
      for (int r = 0; r < 8; ++r)
        memT_s[(mi * 16 + r + hi * 8) * 72 + (njB + j) * 16 + l15] = f2bf(memAcc[j][r]);
    __syncthreads();

    // ---- z += colsum(sk); reset rowsum for next segment ----
    if (tid < 64) {
      float zsum = 0.f;
      for (int s2 = 0; s2 < 64; ++s2) zsum += bf2f(skT_s[tid * 72 + s2]);
      z_s[tid] += zsum;
      rowsum_s[tid] = 0.f;
    }
    __syncthreads();
  }
}

// ---------------------------------------------------------------------------
extern "C" void kernel_launch(void* const* d_in, const int* in_sizes, int n_in,
                              void* d_out, int out_size, void* d_ws, size_t ws_size,
                              hipStream_t stream)
{
  const float* x    = (const float*)d_in[0];
  const float* Wq   = (const float*)d_in[1];
  const float* bq   = (const float*)d_in[2];
  const float* Wk   = (const float*)d_in[3];
  const float* bk   = (const float*)d_in[4];
  const float* Wv   = (const float*)d_in[5];
  const float* bv   = (const float*)d_in[6];
  const float* Wo   = (const float*)d_in[7];
  const float* bo   = (const float*)d_in[8];
  const float* beta = (const float*)d_in[9];
  float* out = (float*)d_out;

  const long long perB = (long long)E_DIM * E_DIM;          // 1M floats per batch
  float* q_ws = (float*)d_ws;
  float* k_ws = q_ws + (long long)B_DIM * perB;
  float* v_ws = k_ws + (long long)B_DIM * perB;
  float* a_ws = v_ws + (long long)B_DIM * perB;             // total 128 MB in d_ws

  dim3 gg(E_DIM / GBN, E_DIM / GBM, B_DIM);
  dim3 bt(256);

  wmma_gemm_abt_bias<<<gg, bt, 0, stream>>>(x, (long long)E_DIM * S_DIM, Wq, bq,
                                            q_ws, perB, S_DIM, E_DIM, S_DIM);
  wmma_gemm_abt_bias<<<gg, bt, 0, stream>>>(x, (long long)E_DIM * S_DIM, Wk, bk,
                                            k_ws, perB, S_DIM, E_DIM, S_DIM);
  wmma_gemm_abt_bias<<<gg, bt, 0, stream>>>(x, (long long)E_DIM * S_DIM, Wv, bv,
                                            v_ws, perB, S_DIM, E_DIM, S_DIM);

  infini_attention_seg<<<dim3(B_DIM * H_DIM), bt, 0, stream>>>(q_ws, k_ws, v_ws, beta, a_ws);

  wmma_gemm_abt_bias<<<gg, bt, 0, stream>>>(a_ws, perB, Wo, bo,
                                            out, perB, E_DIM, E_DIM, E_DIM);
}
